// LocalBlock_63187558859377
// MI455X (gfx1250) — matrix operand order repaired and checked by reading, not compiled, not run on hardware
//
#include <hip/hip_runtime.h>
#include <hip/hip_bf16.h>

// ---------------------------------------------------------------------------
// MI455X (gfx1250) transformer block:
//   LN1 -> QKV GEMM -> local-window flash attention -> out-proj(+residual)
//   -> LN2 -> MLP1(+GELU) -> MLP2(+residual)
// All matmuls run on v_wmma_f32_16x16x32_f16 (wave32 WMMA), f32 accumulate.
// GEMM: 128x64 block tile, 4 waves x (32x64) each, double-buffered LDS,
// one barrier per K-step, global_prefetch_b8 warming K+2.
// ---------------------------------------------------------------------------

typedef __attribute__((ext_vector_type(16))) _Float16 v16h;
typedef __attribute__((ext_vector_type(8)))  _Float16 v8h;
typedef __attribute__((ext_vector_type(8)))  float    v8f;

#define S_LEN   4096
#define D_MODEL 1024
#define NHEAD   16
#define HDIM    64
#define WINDOW  256

static __device__ inline v16h join16(v8h lo, v8h hi) {
  v16h r;
#pragma unroll
  for (int j = 0; j < 8; ++j) { r[j] = lo[j]; r[j + 8] = hi[j]; }
  return r;
}

static __device__ inline v8f zero8() {
  v8f z;
#pragma unroll
  for (int i = 0; i < 8; ++i) z[i] = 0.0f;
  return z;
}

// ---------------------------------------------------------------------------
// fp32 -> f16 cast (weights), grid-stride
// ---------------------------------------------------------------------------
__global__ void cast_f32_f16(const float* __restrict__ src,
                             _Float16* __restrict__ dst, int n) {
  int i = blockIdx.x * blockDim.x + threadIdx.x;
  int stride = gridDim.x * blockDim.x;
  for (; i < n; i += stride) dst[i] = (_Float16)src[i];
}

// ---------------------------------------------------------------------------
// LayerNorm over D=1024, one block (256 threads) per row, fp32 in, f16 out
// ---------------------------------------------------------------------------
static __device__ inline float blockSum256(float v, float* red) {
#pragma unroll
  for (int off = 16; off > 0; off >>= 1) v += __shfl_xor(v, off, 32);
  int w = threadIdx.x >> 5;
  __syncthreads();
  if ((threadIdx.x & 31) == 0) red[w] = v;
  __syncthreads();
  float t = 0.0f;
#pragma unroll
  for (int i = 0; i < 8; ++i) t += red[i];
  return t;
}

__global__ __launch_bounds__(256) void layernorm_to_f16(
    const float* __restrict__ x, const float* __restrict__ w,
    const float* __restrict__ b, _Float16* __restrict__ out) {
  __shared__ float red[8];
  const float* xr = x + (size_t)blockIdx.x * D_MODEL;
  float v[4];
  float s = 0.0f;
#pragma unroll
  for (int i = 0; i < 4; ++i) {
    v[i] = xr[threadIdx.x + i * 256];
    s += v[i];
  }
  float mu = blockSum256(s, red) * (1.0f / D_MODEL);
  float s2 = 0.0f;
#pragma unroll
  for (int i = 0; i < 4; ++i) {
    float d = v[i] - mu;
    s2 += d * d;
  }
  float var = blockSum256(s2, red) * (1.0f / D_MODEL);
  float inv = rsqrtf(var + 1e-5f);
  _Float16* orow = out + (size_t)blockIdx.x * D_MODEL;
#pragma unroll
  for (int i = 0; i < 4; ++i) {
    int c = threadIdx.x + i * 256;
    orow[c] = (_Float16)((v[i] - mu) * inv * w[c] + b[c]);
  }
}

// ---------------------------------------------------------------------------
// Tiled WMMA GEMM: C[M,N] = A[M,K](f16) * B[K,N](f16) + bias
//   block = 128 threads (4 waves), block tile 128x64, K-step 32
//   wave w computes rows [w*32, w*32+32) x 64 cols -> 8 accumulators (v8f)
//   double-buffered LDS: one __syncthreads per K-step; prefetch K+2 tile
// MODE 0: out f16              MODE 1: out f16 after exact GELU
// MODE 2: out f32 = residual + C + bias
// ---------------------------------------------------------------------------
template <int MODE>
__global__ __launch_bounds__(128) void gemm_wmma(
    const _Float16* __restrict__ A, const _Float16* __restrict__ B,
    const float* __restrict__ bias, const float* __restrict__ residual,
    float* __restrict__ outF, _Float16* __restrict__ outH,
    int M, int N, int K) {
  __shared__ _Float16 As[2][128][32];  // row-major A tiles (double buffered)
  __shared__ _Float16 Bt[2][64][32];   // B tiles transposed: [n][k]

  const int tid  = threadIdx.x;
  const int wav  = tid >> 5;
  const int lane = tid & 31;
  const int hb   = lane >> 4;
  const int l15  = lane & 15;
  const int m0   = blockIdx.y * 128;
  const int n0   = blockIdx.x * 64;

  v8f acc[2][4];
#pragma unroll
  for (int u = 0; u < 2; ++u)
#pragma unroll
    for (int c = 0; c < 4; ++c) acc[u][c] = zero8();

  auto loadTile = [&](int buf, int k0) {
    // A tile: 128x32 halves = 512 uint4 chunks; 4 per thread
#pragma unroll
    for (int i = 0; i < 4; ++i) {
      int chunk = tid + i * 128;
      int row = chunk >> 2;
      int col = (chunk & 3) * 8;
      *(uint4*)&As[buf][row][col] =
          *(const uint4*)&A[(size_t)(m0 + row) * K + k0 + col];
    }
    // B tile transposed: 32x64 -> Bt[n][k]; pairs along n, 8 per thread
#pragma unroll
    for (int i = 0; i < 8; ++i) {
      int id = tid + i * 128;
      int kk = id >> 5;
      int nn = (id & 31) * 2;
      const _Float16* bp = &B[(size_t)(k0 + kk) * N + n0 + nn];
      Bt[buf][nn][kk]     = bp[0];
      Bt[buf][nn + 1][kk] = bp[1];
    }
    // Warm L2 for the tile after next (speculative gfx1250 prefetch)
    if (k0 + 32 < K) {
      __builtin_prefetch(&A[(size_t)(m0 + tid) * K + k0 + 32], 0, 0);
      __builtin_prefetch(&B[(size_t)(k0 + 32 + (tid >> 2)) * N + n0], 0, 0);
    }
  };

  const int nK = K >> 5;
  loadTile(0, 0);

  for (int kt = 0; kt < nK; ++kt) {
    const int cur = kt & 1;
    __syncthreads();  // cur tile visible; prior reads of cur^1 complete
    if (kt + 1 < nK) loadTile(cur ^ 1, (kt + 1) * 32);

    // A fragments (16x32, ISA layout: lanes 0-15 K0-7/K16-23; 16-31 K8-15/K24-31)
    v16h af[2];
#pragma unroll
    for (int u = 0; u < 2; ++u) {
      int mr = wav * 32 + u * 16 + l15;
      af[u] = join16(*(const v8h*)&As[cur][mr][8 * hb],
                     *(const v8h*)&As[cur][mr][16 + 8 * hb]);
    }
#pragma unroll
    for (int c = 0; c < 4; ++c) {
      int nn = c * 16 + l15;
      // B fragment (32x16): lane n fixed, k = 8*hb+j / 16+8*hb+j in Bt[n][.]
      v16h bf = join16(*(const v8h*)&Bt[cur][nn][8 * hb],
                       *(const v8h*)&Bt[cur][nn][16 + 8 * hb]);
      acc[0][c] = __builtin_amdgcn_wmma_f32_16x16x32_f16(
          false, af[0], false, bf, (short)0, acc[0][c], false, false);
      acc[1][c] = __builtin_amdgcn_wmma_f32_16x16x32_f16(
          false, af[1], false, bf, (short)0, acc[1][c], false, false);
    }
  }

  // Epilogue (C layout: reg r -> row r + 8*hb, col = l15)
#pragma unroll
  for (int u = 0; u < 2; ++u) {
#pragma unroll
    for (int c = 0; c < 4; ++c) {
      int col = n0 + c * 16 + l15;
      float bv = bias[col];
#pragma unroll
      for (int r = 0; r < 8; ++r) {
        int row = m0 + wav * 32 + u * 16 + r + 8 * hb;
        float v = acc[u][c][r] + bv;
        size_t idx = (size_t)row * N + col;
        if (MODE == 0) {
          outH[idx] = (_Float16)v;
        } else if (MODE == 1) {
          float g = 0.5f * v * (1.0f + erff(v * 0.70710678f));
          outH[idx] = (_Float16)g;
        } else {
          outF[idx] = residual[idx] + v;
        }
      }
    }
  }
}

// ---------------------------------------------------------------------------
// Local-window flash attention. One wave per (head, 16-query tile).
// qkv: f16 [S, 3*D], q cols [0,1024), k [1024,2048), v [2048,3072)
// out: f16 [S, D] attended
// ---------------------------------------------------------------------------
__global__ __launch_bounds__(32) void attn_local(
    const _Float16* __restrict__ qkv, _Float16* __restrict__ attnout) {
  __shared__ _Float16 Qs[16][64];
  __shared__ _Float16 Ks[32][64];
  __shared__ _Float16 Vs[32][64];
  __shared__ _Float16 Ps[16][32];

  const int lane = threadIdx.x;
  const int hb   = lane >> 4;
  const int l15  = lane & 15;
  const int head = blockIdx.y;
  const int q0   = blockIdx.x * 16;
  const int ldq  = 3 * D_MODEL;

  const _Float16* qbase = qkv + head * HDIM;
  const _Float16* kbase = qbase + D_MODEL;
  const _Float16* vbase = qbase + 2 * D_MODEL;

  // Stage Q tile (16 rows x 64 halves); each lane: half a row
  {
    int row = lane >> 1, part = lane & 1;
#pragma unroll
    for (int i = 0; i < 4; ++i) {
      int col = part * 32 + i * 8;
      *(uint4*)&Qs[row][col] =
          *(const uint4*)&qbase[(size_t)(q0 + row) * ldq + col];
    }
  }
  __syncthreads();

  // Q A-fragments for d=0..31 and d=32..63
  v16h qa0 = join16(*(const v8h*)&Qs[l15][8 * hb],
                    *(const v8h*)&Qs[l15][16 + 8 * hb]);
  v16h qa1 = join16(*(const v8h*)&Qs[l15][32 + 8 * hb],
                    *(const v8h*)&Qs[l15][48 + 8 * hb]);

  v8f o[4];
#pragma unroll
  for (int c = 0; c < 4; ++c) o[c] = zero8();
  float mrow[8], lrow[8];
#pragma unroll
  for (int r = 0; r < 8; ++r) { mrow[r] = -1e30f; lrow[r] = 0.0f; }

  int kend = q0 + 15;
  int kstart = q0 - WINDOW;
  if (kstart < 0) kstart = 0;
  kstart &= ~31;

  for (int kt = kstart; kt <= kend; kt += 32) {
    // Stage K/V tiles: lane loads row (key = kt+lane), 64 halves each
    {
      const _Float16* kp = &kbase[(size_t)(kt + lane) * ldq];
      const _Float16* vp = &vbase[(size_t)(kt + lane) * ldq];
#pragma unroll
      for (int i = 0; i < 8; ++i) {
        *(uint4*)&Ks[lane][i * 8] = *(const uint4*)&kp[i * 8];
        *(uint4*)&Vs[lane][i * 8] = *(const uint4*)&vp[i * 8];
      }
    }
    __syncthreads();

    // Scores: two 16x16 tiles, each 2 WMMAs over head_dim (64 = 2x32)
    v8f s[2];
#pragma unroll
    for (int t = 0; t < 2; ++t) {
      s[t] = zero8();
      int krow = t * 16 + l15;  // key row within 32-tile (B lane = key)
      v16h b0 = join16(*(const v8h*)&Ks[krow][8 * hb],
                       *(const v8h*)&Ks[krow][16 + 8 * hb]);
      s[t] = __builtin_amdgcn_wmma_f32_16x16x32_f16(
          false, qa0, false, b0, (short)0, s[t], false, false);
      v16h b1 = join16(*(const v8h*)&Ks[krow][32 + 8 * hb],
                       *(const v8h*)&Ks[krow][48 + 8 * hb]);
      s[t] = __builtin_amdgcn_wmma_f32_16x16x32_f16(
          false, qa1, false, b1, (short)0, s[t], false, false);
    }

    // Mask + online softmax (rows live per-reg; reduce across 16-lane halves)
    float alpha[8];
#pragma unroll
    for (int r = 0; r < 8; ++r) {
      int row = q0 + r + 8 * hb;
      int d0 = row - (kt + l15);
      int d1 = row - (kt + 16 + l15);
      float x0 = (d0 >= 0 && d0 <= WINDOW) ? s[0][r] * 0.125f : -1e30f;
      float x1 = (d1 >= 0 && d1 <= WINDOW) ? s[1][r] * 0.125f : -1e30f;
      float mx = fmaxf(x0, x1);
#pragma unroll
      for (int off = 8; off > 0; off >>= 1)
        mx = fmaxf(mx, __shfl_xor(mx, off, 32));
      float mnew = fmaxf(mrow[r], mx);
      float a = __expf(mrow[r] - mnew);
      float p0 = __expf(x0 - mnew);
      float p1 = __expf(x1 - mnew);
      float rs = p0 + p1;
#pragma unroll
      for (int off = 8; off > 0; off >>= 1) rs += __shfl_xor(rs, off, 32);
      mrow[r] = mnew;
      alpha[r] = a;
      lrow[r] = lrow[r] * a + rs;
      Ps[r + 8 * hb][l15]      = (_Float16)p0;
      Ps[r + 8 * hb][16 + l15] = (_Float16)p1;
    }
    __syncthreads();

    // Rescale running output
#pragma unroll
    for (int c = 0; c < 4; ++c)
#pragma unroll
      for (int r = 0; r < 8; ++r) o[c][r] *= alpha[r];

    // P A-fragment (16x32) from LDS
    v16h pfrag = join16(*(const v8h*)&Ps[l15][16 * hb],
                        *(const v8h*)&Ps[l15][16 * hb + 8]);

    // O += P @ V  (4 output 16-col chunks)
#pragma unroll
    for (int c = 0; c < 4; ++c) {
      v16h vfrag;
#pragma unroll
      for (int j = 0; j < 16; ++j)
        vfrag[j] = Vs[16 * hb + j][c * 16 + l15];
      o[c] = __builtin_amdgcn_wmma_f32_16x16x32_f16(
          false, pfrag, false, vfrag, (short)0, o[c], false, false);
    }
    __syncthreads();
  }

  // Normalize and store attended (f16, [S, D])
#pragma unroll
  for (int r = 0; r < 8; ++r) {
    float inv = (lrow[r] > 0.0f) ? (1.0f / lrow[r]) : 0.0f;
    int row = q0 + r + 8 * hb;
    _Float16* orow = attnout + (size_t)row * D_MODEL + head * HDIM;
#pragma unroll
    for (int c = 0; c < 4; ++c)
      orow[c * 16 + l15] = (_Float16)(o[c][r] * inv);
  }
}

// ---------------------------------------------------------------------------
// Host-side orchestration
// ---------------------------------------------------------------------------
extern "C" void kernel_launch(void* const* d_in, const int* in_sizes, int n_in,
                              void* d_out, int out_size, void* d_ws,
                              size_t ws_size, hipStream_t stream) {
  const float* hs     = (const float*)d_in[0];
  const float* ln1_w  = (const float*)d_in[1];
  const float* ln1_b  = (const float*)d_in[2];
  const float* qkv_w  = (const float*)d_in[3];
  const float* qkv_b  = (const float*)d_in[4];
  const float* out_w  = (const float*)d_in[5];
  const float* out_b  = (const float*)d_in[6];
  const float* ln2_w  = (const float*)d_in[7];
  const float* ln2_b  = (const float*)d_in[8];
  const float* mlp1_w = (const float*)d_in[9];
  const float* mlp1_b = (const float*)d_in[10];
  const float* mlp2_w = (const float*)d_in[11];
  const float* mlp2_b = (const float*)d_in[12];

  // Workspace layout (all offsets 256B aligned)
  char* ws = (char*)d_ws;
  size_t off = 0;
  auto alloc = [&](size_t bytes) {
    void* p = ws + off;
    off += (bytes + 255) & ~(size_t)255;
    return p;
  };
  _Float16* wqkv  = (_Float16*)alloc((size_t)D_MODEL * 3 * D_MODEL * 2);
  _Float16* wout  = (_Float16*)alloc((size_t)D_MODEL * D_MODEL * 2);
  _Float16* wm1   = (_Float16*)alloc((size_t)D_MODEL * 4 * D_MODEL * 2);
  _Float16* wm2   = (_Float16*)alloc((size_t)4 * D_MODEL * D_MODEL * 2);
  _Float16* norm1 = (_Float16*)alloc((size_t)S_LEN * D_MODEL * 2);
  _Float16* qkvb  = (_Float16*)alloc((size_t)S_LEN * 3 * D_MODEL * 2);
  _Float16* attnb = (_Float16*)alloc((size_t)S_LEN * D_MODEL * 2);
  float*    hbuf  = (float*)   alloc((size_t)S_LEN * D_MODEL * 4);
  _Float16* norm2 = (_Float16*)alloc((size_t)S_LEN * D_MODEL * 2);
  _Float16* actb  = (_Float16*)alloc((size_t)S_LEN * 4 * D_MODEL * 2);

  // 1) Weight casts fp32 -> f16
  cast_f32_f16<<<2048, 256, 0, stream>>>(qkv_w, wqkv, D_MODEL * 3 * D_MODEL);
  cast_f32_f16<<<2048, 256, 0, stream>>>(out_w, wout, D_MODEL * D_MODEL);
  cast_f32_f16<<<2048, 256, 0, stream>>>(mlp1_w, wm1, D_MODEL * 4 * D_MODEL);
  cast_f32_f16<<<2048, 256, 0, stream>>>(mlp2_w, wm2, 4 * D_MODEL * D_MODEL);

  // 2) LN1
  layernorm_to_f16<<<S_LEN, 256, 0, stream>>>(hs, ln1_w, ln1_b, norm1);

  // 3) QKV GEMM: [4096,1024] x [1024,3072] -> f16
  gemm_wmma<0><<<dim3(3 * D_MODEL / 64, S_LEN / 128), 128, 0, stream>>>(
      norm1, wqkv, qkv_b, nullptr, nullptr, qkvb, S_LEN, 3 * D_MODEL, D_MODEL);

  // 4) Local-window attention
  attn_local<<<dim3(S_LEN / 16, NHEAD), 32, 0, stream>>>(qkvb, attnb);

  // 5) Out-proj + residual: h = hs + attn @ out_w + out_b (f32)
  gemm_wmma<2><<<dim3(D_MODEL / 64, S_LEN / 128), 128, 0, stream>>>(
      attnb, wout, out_b, hs, hbuf, nullptr, S_LEN, D_MODEL, D_MODEL);

  // 6) LN2
  layernorm_to_f16<<<S_LEN, 256, 0, stream>>>(hbuf, ln2_w, ln2_b, norm2);

  // 7) MLP1 + GELU: [4096,1024] x [1024,4096] -> f16
  gemm_wmma<1><<<dim3(4 * D_MODEL / 64, S_LEN / 128), 128, 0, stream>>>(
      norm2, wm1, mlp1_b, nullptr, nullptr, actb, S_LEN, 4 * D_MODEL, D_MODEL);

  // 8) MLP2 + residual: out = h + act @ mlp2_w + mlp2_b (f32)
  gemm_wmma<2><<<dim3(D_MODEL / 64, S_LEN / 128), 128, 0, stream>>>(
      actb, wm2, mlp2_b, hbuf, (float*)d_out, nullptr, S_LEN, D_MODEL,
      4 * D_MODEL);
}
